// GNNEncoder_81801947120311
// MI455X (gfx1250) — matrix operand order repaired and checked by reading, not compiled, not run on hardware
//
#include <hip/hip_runtime.h>
#include <hip/hip_bf16.h>
#include <math.h>

typedef __attribute__((ext_vector_type(16))) _Float16 v16h;
typedef __attribute__((ext_vector_type(8)))  float    v8f;

#define D_DIM 64
#define EPS_LN 1e-5f
#define SCALE_ATT 0.125f   // 1/sqrt(64)

// ---------------------------------------------------------------------------
// One-time weight pack: f32 row-major (K x N) -> f16 WMMA B-fragment order
//   Wp[layer][mat(4)][kstep(2)][ntile(4)][lane(32)][elem(16)]
// Fragment element j of lane l (ncol = nt*16 + (l&15)):
//   K = kstep*32 + (l>>4)*16 + j
// Total: L*4*2*4*32*16 = 49152 f16 (96 KB) -- trivial one-shot kernel.
// ---------------------------------------------------------------------------
__global__ __launch_bounds__(256)
void pack_weights_kernel(const float* __restrict__ Wq, const float* __restrict__ Wk,
                         const float* __restrict__ Wv, const float* __restrict__ Ws,
                         _Float16* __restrict__ Wp, int nLayers)
{
    int idx = blockIdx.x * blockDim.x + threadIdx.x;
    int total = nLayers * 4 * 2 * 4 * 32 * 16;   // = nLayers << 14
    if (idx >= total) return;
    int j    = idx & 15;
    int lane = (idx >> 4) & 31;
    int nt   = (idx >> 9) & 3;
    int ks2  = (idx >> 11) & 1;
    int mat  = (idx >> 12) & 3;
    int l    = idx >> 14;
    const float* W = (mat == 0) ? Wq : (mat == 1) ? Wk : (mat == 2) ? Wv : Ws;
    W += (size_t)l * D_DIM * D_DIM;
    int ncol = nt * 16 + (lane & 15);
    int kk   = ks2 * 32 + (lane >> 4) * 16 + j;
    Wp[idx] = (_Float16)W[(size_t)kk * D_DIM + ncol];
}

// ---------------------------------------------------------------------------
// Fused projection GEMM: Y[n, 0:256] = [x@Wq+bq | x@Wk+bk | x@Wv+bv | x@Ws+bs]
// block = 128 threads (4 waves), one 16-node strip per block; wave w owns
// matrix w and issues 4 tiles x 2 K-steps = 8 v_wmma_f32_16x16x32_f16.
// A: two contiguous float4-pairs per lane per K-step; B: one 32B v16h per lane.
// ---------------------------------------------------------------------------
__global__ __launch_bounds__(128)
void gemm_qkvs_kernel(const float* __restrict__ X,
                      const _Float16* __restrict__ Wp,   // this layer's 4 packed mats
                      const float* __restrict__ bq, const float* __restrict__ bk,
                      const float* __restrict__ bv, const float* __restrict__ bs,
                      float* __restrict__ Y, int nNodes)
{
    const int wave = threadIdx.x >> 5;      // 0..3 -> q,k,v,s
    const int lane = threadIdx.x & 31;
    const int mbase = blockIdx.x * 16;
    if (mbase >= nNodes) return;

    const float* B = (wave == 0) ? bq : (wave == 1) ? bk : (wave == 2) ? bv : bs;
    const _Float16* wp = Wp + (size_t)wave * (2 * 4 * 32 * 16);

    const int mrow = lane & 15;
    const int half = lane >> 4;
    int xr = mbase + mrow;
    if (xr >= nNodes) xr = nNodes - 1;
    const float4* x4 = (const float4*)(X + (size_t)xr * D_DIM);

    v8f acc0 = {}, acc1 = {}, acc2 = {}, acc3 = {};

    #pragma unroll
    for (int ks2 = 0; ks2 < 2; ++ks2) {
        // --- A fragment: elems 0..7 -> K = ks2*32 + half*16 + (0..7)
        //                 elems 8..15 -> K = ks2*32 + half*16 + 8 + (0..7)
        const int kA0 = ks2 * 32 + half * 16;
        const int kA1 = kA0 + 8;
        float4 r0 = x4[kA0 >> 2], r1 = x4[(kA0 >> 2) + 1];
        float4 r2 = x4[kA1 >> 2], r3 = x4[(kA1 >> 2) + 1];
        v16h a;
        a[0]  = (_Float16)r0.x; a[1]  = (_Float16)r0.y;
        a[2]  = (_Float16)r0.z; a[3]  = (_Float16)r0.w;
        a[4]  = (_Float16)r1.x; a[5]  = (_Float16)r1.y;
        a[6]  = (_Float16)r1.z; a[7]  = (_Float16)r1.w;
        a[8]  = (_Float16)r2.x; a[9]  = (_Float16)r2.y;
        a[10] = (_Float16)r2.z; a[11] = (_Float16)r2.w;
        a[12] = (_Float16)r3.x; a[13] = (_Float16)r3.y;
        a[14] = (_Float16)r3.z; a[15] = (_Float16)r3.w;

        #pragma unroll
        for (int nt = 0; nt < 4; ++nt) {
            // packed fragment: one contiguous 32B load per lane
            v16h b = *(const v16h*)(wp + (((size_t)ks2 * 4 + nt) * 32 + lane) * 16);
            v8f c = (nt == 0) ? acc0 : (nt == 1) ? acc1 : (nt == 2) ? acc2 : acc3;
            c = __builtin_amdgcn_wmma_f32_16x16x32_f16(false, a, false, b,
                                                       (short)0, c, false, false);
            if      (nt == 0) acc0 = c;
            else if (nt == 1) acc1 = c;
            else if (nt == 2) acc2 = c;
            else              acc3 = c;
        }
    }

    // --- store D tiles: VGPR r -> row M = half*8 + r, col = lane&15 ---
    #pragma unroll
    for (int nt = 0; nt < 4; ++nt) {
        const int ncol = nt * 16 + (lane & 15);
        const float bias = B[ncol];
        v8f c = (nt == 0) ? acc0 : (nt == 1) ? acc1 : (nt == 2) ? acc2 : acc3;
        #pragma unroll
        for (int r = 0; r < 8; ++r) {
            int m = mbase + half * 8 + r;
            if (m < nNodes)
                Y[(size_t)m * 256 + wave * 64 + ncol] = c[r] + bias;
        }
    }
}

// ---------------------------------------------------------------------------
__global__ __launch_bounds__(256)
void init_buffers_kernel(float* __restrict__ agg, float* __restrict__ denom,
                         unsigned* __restrict__ mmax, int nNodes)
{
    int i = blockIdx.x * blockDim.x + threadIdx.x;
    if (i < nNodes * D_DIM) agg[i] = 0.0f;
    if (i < nNodes) { denom[i] = 0.0f; mmax[i] = 0u; }
}

// ---------------------------------------------------------------------------
// score[e] = <q[dst], k[src]> * SCALE; segment max via order-preserving u32.
// 8 lanes per edge, each lane reads 2x float4 of q and k (rows 256B aligned).
// ---------------------------------------------------------------------------
__global__ __launch_bounds__(256)
void edge_score_kernel(const float* __restrict__ Y,
                       const int* __restrict__ src, const int* __restrict__ dst,
                       float* __restrict__ score, unsigned* __restrict__ mmax,
                       int nEdges)
{
    int g = blockIdx.x * blockDim.x + threadIdx.x;
    int e = g >> 3;
    int sub = g & 7;
    if (e >= nEdges) return;
    int s = src[e], d = dst[e];
    const float4* q = (const float4*)(Y + (size_t)d * 256);        // cols 0..63
    const float4* k = (const float4*)(Y + (size_t)s * 256 + 64);   // cols 64..127
    float4 qa = q[sub * 2], qb = q[sub * 2 + 1];
    float4 ka = k[sub * 2], kb = k[sub * 2 + 1];
    float p = qa.x * ka.x + qa.y * ka.y + qa.z * ka.z + qa.w * ka.w
            + qb.x * kb.x + qb.y * kb.y + qb.z * kb.z + qb.w * kb.w;
    p += __shfl_xor(p, 1, 32);
    p += __shfl_xor(p, 2, 32);
    p += __shfl_xor(p, 4, 32);
    if (sub == 0) {
        p *= SCALE_ATT;
        score[e] = p;
        unsigned bits = __float_as_uint(p);
        unsigned ord  = (bits & 0x80000000u) ? ~bits : (bits | 0x80000000u);
        atomicMax(&mmax[d], ord);
    }
}

// ---------------------------------------------------------------------------
// Fused softmax-numerator + aggregation: one wave per edge.
//   ev = exp(score - m[dst]);  denom[dst] += ev (lane 0);
//   agg[dst][c] += ev * v[src][c]  (2 cols/lane)
// Division by denom is deferred to the finalize kernel.
// ---------------------------------------------------------------------------
__global__ __launch_bounds__(256)
void edge_softmax_aggregate_kernel(const float* __restrict__ Y,
                                   const float* __restrict__ score,
                                   const int* __restrict__ src,
                                   const int* __restrict__ dst,
                                   const unsigned* __restrict__ mmax,
                                   float* __restrict__ denom,
                                   float* __restrict__ agg, int nEdges)
{
    int e = (blockIdx.x * blockDim.x + threadIdx.x) >> 5;
    int lane = threadIdx.x & 31;
    if (e >= nEdges) return;
    int s = src[e], d = dst[e];
    unsigned ord  = mmax[d];
    unsigned bits = (ord & 0x80000000u) ? (ord & 0x7fffffffu) : ~ord;
    float m  = __uint_as_float(bits);
    float ev = __expf(score[e] - m);
    if (lane == 0) atomicAdd(&denom[d], ev);
    const float* v = Y + (size_t)s * 256 + 128;    // cols 128..191
    float* a = agg + (size_t)d * D_DIM;
    atomicAdd(&a[lane],      ev * v[lane]);
    atomicAdd(&a[lane + 32], ev * v[lane + 32]);
}

// ---------------------------------------------------------------------------
// h = relu(agg/denom + skip); layer0: stash x_init; else h += x_init;
// LayerNorm over D=64 with wave32 shuffle reductions. One wave per node.
// ---------------------------------------------------------------------------
__global__ __launch_bounds__(256)
void finalize_ln_kernel(const float* __restrict__ agg, const float* __restrict__ denom,
                        const float* __restrict__ Y, float* __restrict__ x_init,
                        const float* __restrict__ gamma, const float* __restrict__ beta,
                        float* __restrict__ x_out, int nNodes, int layer)
{
    int n = (blockIdx.x * blockDim.x + threadIdx.x) >> 5;
    int lane = threadIdx.x & 31;
    if (n >= nNodes) return;
    float dn  = denom[n];
    float inv = (dn > 0.0f) ? (1.0f / dn) : 0.0f;
    const float* a  = agg + (size_t)n * D_DIM;
    const float* sk = Y + (size_t)n * 256 + 192;   // skip cols 192..255
    float h0 = fmaxf(a[lane]      * inv + sk[lane],      0.0f);
    float h1 = fmaxf(a[lane + 32] * inv + sk[lane + 32], 0.0f);
    if (layer == 0) {
        x_init[(size_t)n * D_DIM + lane]      = h0;
        x_init[(size_t)n * D_DIM + lane + 32] = h1;
    } else {
        h0 += x_init[(size_t)n * D_DIM + lane];
        h1 += x_init[(size_t)n * D_DIM + lane + 32];
    }
    float s = h0 + h1;
    #pragma unroll
    for (int o = 16; o >= 1; o >>= 1) s += __shfl_xor(s, o, 32);
    float mu = s * (1.0f / 64.0f);
    float d0 = h0 - mu, d1 = h1 - mu;
    float vs = d0 * d0 + d1 * d1;
    #pragma unroll
    for (int o = 16; o >= 1; o >>= 1) vs += __shfl_xor(vs, o, 32);
    float rstd = rsqrtf(vs * (1.0f / 64.0f) + EPS_LN);
    x_out[(size_t)n * D_DIM + lane]      = d0 * rstd * gamma[lane]      + beta[lane];
    x_out[(size_t)n * D_DIM + lane + 32] = d1 * rstd * gamma[lane + 32] + beta[lane + 32];
}

// ---------------------------------------------------------------------------
extern "C" void kernel_launch(void* const* d_in, const int* in_sizes, int n_in,
                              void* d_out, int out_size, void* d_ws, size_t ws_size,
                              hipStream_t stream)
{
    const float* x   = (const float*)d_in[0];
    const int*   ei  = (const int*)  d_in[1];
    const float* Wq  = (const float*)d_in[2];
    const float* bq  = (const float*)d_in[3];
    const float* Wk  = (const float*)d_in[4];
    const float* bk  = (const float*)d_in[5];
    const float* Wv  = (const float*)d_in[6];
    const float* bv  = (const float*)d_in[7];
    const float* Ws  = (const float*)d_in[8];
    const float* bs  = (const float*)d_in[9];
    const float* gam = (const float*)d_in[10];
    const float* bet = (const float*)d_in[11];

    const int N = in_sizes[0] / D_DIM;
    const int E = in_sizes[1] / 2;
    const int L = 3;
    const int* srcI = ei;          // edge_index[0]
    const int* dstI = ei + E;      // edge_index[1]

    // workspace carve-up (256B aligned)
    char* ws = (char*)d_ws;
    size_t off = 0;
    auto carve = [&](size_t bytes) -> char* {
        char* p = ws + off;
        off += (bytes + 255) & ~(size_t)255;
        return p;
    };
    float*    Yb     = (float*)   carve((size_t)N * 256 * sizeof(float));
    float*    aggB   = (float*)   carve((size_t)N * D_DIM * sizeof(float));
    float*    xcur   = (float*)   carve((size_t)N * D_DIM * sizeof(float));
    float*    xinit  = (float*)   carve((size_t)N * D_DIM * sizeof(float));
    float*    denomB = (float*)   carve((size_t)N * sizeof(float));
    unsigned* mmaxB  = (unsigned*)carve((size_t)N * sizeof(unsigned));
    float*    scoreB = (float*)   carve((size_t)E * sizeof(float));
    _Float16* WpB    = (_Float16*)carve((size_t)L * 4 * 2 * 4 * 32 * 16 * sizeof(_Float16));
    (void)ws_size;

    const int packThreads = L * 4 * 2 * 4 * 32 * 16;     // L << 14
    const int packBlks    = (packThreads + 255) / 256;
    const int gemmBlocks  = (N + 15) / 16;
    const int initBlocks  = (N * D_DIM + 255) / 256;
    const int scoreBlks   = (E * 8 + 255) / 256;
    const int aggBlks     = (E * 32 + 255) / 256;
    const int nodeBlks    = (N * 32 + 255) / 256;

    pack_weights_kernel<<<packBlks, 256, 0, stream>>>(Wq, Wk, Wv, Ws, WpB, L);

    for (int l = 0; l < L; ++l) {
        const float* xin = (l == 0) ? x : xcur;
        const size_t bo = (size_t)l * D_DIM;
        const _Float16* WpL = WpB + (size_t)l * (4 * 2 * 4 * 32 * 16);

        gemm_qkvs_kernel<<<gemmBlocks, 128, 0, stream>>>(
            xin, WpL, bq + bo, bk + bo, bv + bo, bs + bo, Yb, N);

        init_buffers_kernel<<<initBlocks, 256, 0, stream>>>(aggB, denomB, mmaxB, N);

        edge_score_kernel<<<scoreBlks, 256, 0, stream>>>(Yb, srcI, dstI, scoreB, mmaxB, E);

        edge_softmax_aggregate_kernel<<<aggBlks, 256, 0, stream>>>(
            Yb, scoreB, srcI, dstI, mmaxB, denomB, aggB, E);

        float* xout = (l == L - 1) ? (float*)d_out : xcur;
        finalize_ln_kernel<<<nodeBlks, 256, 0, stream>>>(
            aggB, denomB, Yb, xinit, gam + bo, bet + bo, xout, N, l);
    }
}
